// NeuromorphicLM_80144089743479
// MI455X (gfx1250) — matrix operand order repaired and checked by reading, not compiled, not run on hardware
//
#include <hip/hip_runtime.h>

typedef _Float16 half_t;
typedef __attribute__((ext_vector_type(16))) _Float16 v16h;
typedef __attribute__((ext_vector_type(8)))  float    v8f;

constexpr int kBS = 4, kN = 1024, kVOCAB = 32000, kD = 1024, kC = 16,
              kDCOL = 64, kNB = 8, kM = 64, kL = 2, kE = 128, kTRAIL = 4;

// ---------------- helpers ----------------

__device__ __forceinline__ float sigmoidf_(float x) { return 1.f / (1.f + __expf(-x)); }

template <int NT>
__device__ __forceinline__ float block_reduce_sum(float v, float* red) {
  const int t = threadIdx.x;
  red[t] = v;
  __syncthreads();
#pragma unroll
  for (int s = NT >> 1; s > 0; s >>= 1) {
    if (t < s) red[t] += red[t + s];
    __syncthreads();
  }
  float r = red[0];
  __syncthreads();
  return r;
}

// CDNA5 WMMA f16 fragment loaders (wave32).
// A 16x32 (MxK): lane<16 -> row=lane,   K = 0..15
//                lane>=16 -> row=lane-16, K = 16..31
__device__ __forceinline__ v16h frag_a(const half_t* p, int lane) {
  const int kg = (lane & 16) ? 16 : 0;
  v16h f;
#pragma unroll
  for (int i = 0; i < 16; ++i) f[i] = p[kg + i];
  return f;
}
// B 32x16 (KxN): lane col = lane&15; lane<16 -> K=0..15, lane>=16 -> K=16..31
__device__ __forceinline__ v16h frag_b(const half_t* p, int lane) {
  const int kb = (lane & 16) ? 16 : 0;
  v16h f;
#pragma unroll
  for (int i = 0; i < 16; ++i) f[i] = p[kb + i];
  return f;
}
__device__ __forceinline__ v8f wmma_f16(v16h a, v16h b, v8f c) {
  return __builtin_amdgcn_wmma_f32_16x16x32_f16(false, a, false, b, (short)0, c,
                                                false, false);
}

// CDNA5 async memory->LDS copy (ASYNCcnt-tracked); generic-from-local pointers
// carry the LDS byte offset in the low 32 bits.
__device__ __forceinline__ unsigned lds_off32(const void* p) {
  return (unsigned)(uintptr_t)p;
}
__device__ __forceinline__ void async_load_b128(unsigned lds_dst, const void* gsrc) {
  asm volatile("global_load_async_to_lds_b128 %0, %1, off"
               :
               : "v"(lds_dst), "v"(gsrc)
               : "memory");
}
__device__ __forceinline__ void wait_async0() {
  asm volatile("s_wait_asynccnt 0x0" ::: "memory");
}

// ---------------- prep kernels ----------------

// Wdt[bank][e][d] = (f16) W_delta[bank][d][e]
__global__ void wdt_kernel(const float* __restrict__ Wd, half_t* __restrict__ Wdt) {
  size_t i = (size_t)blockIdx.x * 256 + threadIdx.x;
  int bank = (int)(i / ((size_t)kD * kD));
  int e = (int)((i / kD) % kD);
  int d = (int)(i % kD);
  Wdt[i] = (half_t)Wd[((size_t)bank * kD + d) * kD + e];
}

// per (b,bank,m) row of em_K: f16 copy + normalized f16 copy
__global__ __launch_bounds__(256) void kprep_kernel(const float* __restrict__ emK,
                                                    half_t* __restrict__ kf,
                                                    half_t* __restrict__ knh) {
  __shared__ float red[256];
  const int row = blockIdx.x;  // [0, BS*NB*M)
  const int t = threadIdx.x;
  const size_t base = (size_t)row * kD;
  float ss = 0.f;
  for (int d = t; d < kD; d += 256) {
    float v = emK[base + d];
    ss += v * v;
  }
  float n2 = block_reduce_sum<256>(ss, red);
  float inv = 1.f / (sqrtf(n2) + 1e-6f);
  for (int d = t; d < kD; d += 256) {
    float v = emK[base + d];
    kf[base + d] = (half_t)v;
    knh[base + d] = (half_t)(v * inv);
  }
}

// Vt[b,bank][d][m] = (f16) em_V[b,bank][m][d]
__global__ void vt_kernel(const float* __restrict__ emV, half_t* __restrict__ Vt) {
  size_t i = (size_t)blockIdx.x * 256 + threadIdx.x;  // over BS*NB*D*M
  int bb = (int)(i / ((size_t)kD * kM));
  int d = (int)((i / kM) % kD);
  int m = (int)(i % kM);
  Vt[i] = (half_t)emV[((size_t)bb * kM + m) * kD + d];
}

__global__ void embh_kernel(const float* __restrict__ emb, half_t* __restrict__ embh) {
  size_t i = (size_t)blockIdx.x * 256 + threadIdx.x;
  if (i < (size_t)kVOCAB * kD) embh[i] = (half_t)emb[i];
}

// ---------------- embedding ----------------

__global__ void embed_kernel(const int* __restrict__ ids, const float* __restrict__ emb,
                             const float* __restrict__ pos, float* __restrict__ X,
                             float* __restrict__ XC) {
  const int row = blockIdx.x;  // b*N+n
  const int n = row & (kN - 1);
  const int tok = ids[row];
  for (int d = threadIdx.x; d < kD; d += blockDim.x) {
    float v = emb[(size_t)tok * kD + d] + pos[(size_t)n * kD + d];
    X[(size_t)row * kD + d] = v;
    XC[(size_t)row * kD + d] = v;
  }
}

// ---------------- scan layer (3 phases) ----------------

__global__ __launch_bounds__(128) void scan_phaseA(
    const float* __restrict__ H, const float* __restrict__ Win,
    const float* __restrict__ Wg, const float* __restrict__ g,
    const float* __restrict__ b, float* __restrict__ Z, float* __restrict__ Aa) {
  __shared__ float xs[kDCOL];
  const int bc = blockIdx.x;
  const int c = bc & (kC - 1);
  const int bn = bc >> 4;
  const int t = threadIdx.x;
  const size_t base = ((size_t)bn * kC + c) * kDCOL;
  if (t < kDCOL) xs[t] = H[base + t];
  __syncthreads();
  float mu = 0.f, s2 = 0.f;
#pragma unroll 8
  for (int d = 0; d < kDCOL; ++d) {
    float v = xs[d];
    mu += v;
    s2 += v * v;
  }
  mu *= (1.f / kDCOL);
  float var = s2 * (1.f / kDCOL) - mu * mu;
  float rstd = rsqrtf(var + 1e-5f);
  __syncthreads();
  if (t < kDCOL) xs[t] = (xs[t] - mu) * rstd * g[t] + b[t];
  __syncthreads();
  const float* win = Win + (size_t)c * kDCOL * kE;
  const float* wg = Wg + (size_t)c * kDCOL * kE;
  float z = 0.f, a = 0.f;
#pragma unroll 4
  for (int d = 0; d < kDCOL; ++d) {
    float xv = xs[d];
    z += xv * win[d * kE + t];
    a += xv * wg[d * kE + t];
  }
  a = sigmoidf_(a);
  const size_t zi = ((size_t)bn * kC + c) * kE + t;
  Z[zi] = z;
  Aa[zi] = a;
}

__global__ void scan_phaseB(float* __restrict__ Z, const float* __restrict__ Aa) {
  const int t = blockIdx.x * blockDim.x + threadIdx.x;  // [0, BS*C*E)
  const int e = t & (kE - 1);
  const int c = (t >> 7) & (kC - 1);
  const int b = t >> 11;
  float h = 0.f;
  for (int n = 0; n < kN; ++n) {
    const size_t i = (((size_t)b * kN + n) * kC + c) * kE + e;
    const float a = Aa[i];
    h = a * h + (1.f - a) * Z[i];
    Z[i] = h;
  }
}

__global__ __launch_bounds__(64) void scan_phaseC(float* __restrict__ H,
                                                  const float* __restrict__ Z,
                                                  const float* __restrict__ Wout) {
  __shared__ float hs[kE];
  const int bc = blockIdx.x;
  const int c = bc & (kC - 1);
  const int bn = bc >> 4;
  const int t = threadIdx.x;
  const float* zrow = Z + ((size_t)bn * kC + c) * kE;
  hs[t] = zrow[t];
  hs[t + 64] = zrow[t + 64];
  __syncthreads();
  const float* wo = Wout + (size_t)c * kE * kDCOL;
  float o = 0.f;
#pragma unroll 4
  for (int e = 0; e < kE; ++e) o += hs[e] * wo[e * kDCOL + t];
  H[((size_t)bn * kC + c) * kDCOL + t] += o;
}

// ---------------- PCM + seed/w_cand (fused, per (bn,c)) ----------------

__global__ __launch_bounds__(64) void pcm_kernel(
    float* __restrict__ H, const float* __restrict__ XC,
    const float* __restrict__ Wpred, const float* __restrict__ Wseed,
    const float* __restrict__ Ww, float* __restrict__ surp,
    half_t* __restrict__ surph, float* __restrict__ seed,
    float* __restrict__ wcand, float* __restrict__ aux) {
  __shared__ float hs[kDCOL], h2[kDCOL], red[64];
  const int bc = blockIdx.x;
  const int c = bc & (kC - 1);
  const int bn = bc >> 4;
  const int t = threadIdx.x;
  const size_t base = ((size_t)bn * kC + c) * kDCOL;
  hs[t] = H[base + t];
  __syncthreads();
  const float* wp = Wpred + (size_t)c * kDCOL * kDCOL;
  float zh = 0.f;
#pragma unroll 8
  for (int d = 0; d < kDCOL; ++d) zh += hs[d] * wp[d * kDCOL + t];
  const float x = XC[base + t];
  const float diff = zh - x;
  const float sur = fabsf(diff);
  const float hn = hs[t] * (1.f + tanhf(sur));
  h2[t] = hn;
  H[base + t] = hn;
  surp[base + t] = sur;
  surph[base + t] = (half_t)sur;
  red[t] = diff * diff;
  __syncthreads();
#pragma unroll
  for (int s = 32; s > 0; s >>= 1) {
    if (t < s) red[t] += red[t + s];
    __syncthreads();
  }
  if (t == 0) unsafeAtomicAdd(aux, red[0]);
  const float* wsd = Wseed + (size_t)c * kDCOL * kDCOL;
  const float* wwc = Ww + (size_t)c * kDCOL * kDCOL;
  float sd = 0.f, wc = 0.f;
#pragma unroll 8
  for (int d = 0; d < kDCOL; ++d) {
    const float hv = h2[d];
    sd += hv * wsd[d * kDCOL + t];
    wc += hv * wwc[d * kDCOL + t];
  }
  seed[base + t] = sd;
  wcand[base + t] = wc;
}

// ---------------- novelty (per (b,n) row) ----------------

__global__ __launch_bounds__(256) void novelty_kernel(
    const float* __restrict__ H, const float* __restrict__ wcand,
    const float* __restrict__ surp, const float* __restrict__ Wnw,
    const float* __restrict__ Wnb, const half_t* __restrict__ knh,
    float* __restrict__ novsum) {
  __shared__ float wn[kD];
  __shared__ float red[256];
  __shared__ float sims[kNB * kM];
  __shared__ float nvz[kNB];
  const int bn = blockIdx.x;
  const int b = bn >> 10;
  const int t = threadIdx.x;
  const size_t base = (size_t)bn * kD;
  float ss = 0.f, sm = 0.f;
  for (int d = t; d < kD; d += 256) {
    const float v = wcand[base + d];
    wn[d] = v;
    ss += v * v;
    sm += surp[base + d];
  }
  const float n2 = block_reduce_sum<256>(ss, red);
  const float smsum = block_reduce_sum<256>(sm, red);
  const float inv = 1.f / (sqrtf(n2) + 1e-6f);
  for (int d = t; d < kD; d += 256) wn[d] *= inv;
  __syncthreads();
  const float smag = sigmoidf_(smsum * (1.f / kD));
  float wnv[kNB];
#pragma unroll
  for (int bank = 0; bank < kNB; ++bank) {
    float p = 0.f;
    for (int d = t; d < kD; d += 256) p += H[base + d] * Wnw[(size_t)d * kNB + bank];
    const float tot = block_reduce_sum<256>(p, red);
    wnv[bank] = sigmoidf_(tot + Wnb[bank]);
  }
  for (int p = t; p < kNB * kM; p += 256) {
    const int bank = p >> 6, m = p & (kM - 1);
    const half_t* kr = knh + (((size_t)b * kNB + bank) * kM + m) * kD;
    float acc = 0.f;
#pragma unroll 8
    for (int d = 0; d < kD; ++d) acc += wn[d] * (float)kr[d];
    sims[p] = acc;
  }
  __syncthreads();
  if (t < kNB) {
    float mx = -1e30f;
    for (int m = 0; m < kM; ++m) mx = fmaxf(mx, sims[t * kM + m]);
    nvz[t] = wnv[t] * smag * fmaxf(0.f, 1.f - mx);
  }
  __syncthreads();
  if (t == 0) {
    float s = 0.f;
#pragma unroll
    for (int i = 0; i < kNB; ++i) s += nvz[i];
    novsum[bn] = s;
  }
}

// ---------------- generic NT GEMM with WMMA f16 ----------------
// C[M x N](f32, ld=LDC) = A[M x K](f16 row-major) * Bt[N x K](f16 row-major)^T
// grid = (N/64, M/64), block = 128 (4 waves, each 32x32 of a 64x64 tile).
// Tiles staged with CDNA5 async memory->LDS copies (ASYNCcnt).
template <int LDC>
__global__ __launch_bounds__(128) void gemm_nt_kernel(const half_t* __restrict__ A,
                                                      const half_t* __restrict__ Bt,
                                                      float* __restrict__ Cc, int K) {
  __shared__ half_t As[64][32];
  __shared__ half_t Bs[64][32];
  const int tid = threadIdx.x;
  const int lane = tid & 31;
  const int wave = tid >> 5;
  const int wx = wave & 1, wy = wave >> 1;
  const int m0 = blockIdx.y * 64, n0 = blockIdx.x * 64;
  v8f acc[2][2] = {};
  const int lr = tid >> 1;
  const int lc = (tid & 1) * 16;
  // loop-invariant LDS destinations + per-row global bases
  const unsigned la0 = lds_off32(&As[lr][lc]);
  const unsigned la1 = lds_off32(&As[lr][lc + 8]);
  const unsigned lb0 = lds_off32(&Bs[lr][lc]);
  const unsigned lb1 = lds_off32(&Bs[lr][lc + 8]);
  const half_t* ga = A + (size_t)(m0 + lr) * K + lc;
  const half_t* gb = Bt + (size_t)(n0 + lr) * K + lc;
  for (int k0 = 0; k0 < K; k0 += 32) {
    async_load_b128(la0, ga + k0);
    async_load_b128(la1, ga + k0 + 8);
    async_load_b128(lb0, gb + k0);
    async_load_b128(lb1, gb + k0 + 8);
    wait_async0();
    __syncthreads();
    v16h aF[2], bF[2];
#pragma unroll
    for (int ti = 0; ti < 2; ++ti)
      aF[ti] = frag_a(&As[wy * 32 + ti * 16 + (lane & 15)][0], lane);
#pragma unroll
    for (int tj = 0; tj < 2; ++tj)
      bF[tj] = frag_b(&Bs[wx * 32 + tj * 16 + (lane & 15)][0], lane);
#pragma unroll
    for (int ti = 0; ti < 2; ++ti)
#pragma unroll
      for (int tj = 0; tj < 2; ++tj) acc[ti][tj] = wmma_f16(aF[ti], bF[tj], acc[ti][tj]);
    __syncthreads();
  }
  const int rbase = (lane & 16) ? 8 : 0;
  const int cl = lane & 15;
#pragma unroll
  for (int ti = 0; ti < 2; ++ti)
#pragma unroll
    for (int tj = 0; tj < 2; ++tj) {
      float* crow = Cc + (size_t)(m0 + wy * 32 + ti * 16 + rbase) * LDC +
                    (n0 + wx * 32 + tj * 16 + cl);
#pragma unroll
      for (int r = 0; r < 8; ++r) crow[(size_t)r * LDC] = acc[ti][tj][r];
    }
}

// ---------------- pm cumsum + gated read (per bank) ----------------

__global__ void pm_cum_kernel(const float* __restrict__ tmp, const float* __restrict__ H,
                              const float* __restrict__ pmstate,
                              float* __restrict__ integ, int bank) {
  const int t = blockIdx.x * blockDim.x + threadIdx.x;  // [0, BS*D)
  const int s = t >> 10;
  const int e = t & (kD - 1);
  const float ps = pmstate[((size_t)s * kNB + bank) * kD + e];
  float cum = 0.f;
  for (int n = 0; n < kN; ++n) {
    const size_t i = ((size_t)s * kN + n) * kD + e;
    cum += tmp[i];
    integ[i] += H[i] * sigmoidf_(ps + cum);
  }
}

// ---------------- episodic cumulative write (bank-summed) ----------------

__global__ void em_cum_kernel(const float* __restrict__ novs,
                              const float* __restrict__ wcand,
                              float* __restrict__ integ) {
  const int t = blockIdx.x * blockDim.x + threadIdx.x;  // [0, BS*D)
  const int s = t >> 10;
  const int e = t & (kD - 1);
  float cum = 0.f;
  for (int n = 0; n < kN; ++n) {
    const size_t i = ((size_t)s * kN + n) * kD + e;
    cum += novs[s * kN + n] * wcand[i];
    integ[i] += cum;
  }
}

// ---------------- trail attention (WMMA, one wave per 16 rows x bank) ----------------

__global__ __launch_bounds__(32) void trail_attn_kernel(const float* __restrict__ seed,
                                                        const half_t* __restrict__ Kf,
                                                        const half_t* __restrict__ Vt,
                                                        float* __restrict__ integ) {
  __shared__ half_t Qh[16][kD];     // 32 KB
  __shared__ half_t attnh[16][kM];  // 2 KB
  __shared__ float sc[16][kM];      // 4 KB
  const int lane = threadIdx.x;
  const int n0 = blockIdx.x * 16;
  const int bank = blockIdx.y;
  const int b = blockIdx.z;
  const size_t bb = (size_t)b * kNB + bank;
  for (int idx = lane; idx < 16 * kD; idx += 32) {
    const int r = idx >> 10, d = idx & (kD - 1);
    Qh[r][d] = (half_t)seed[((size_t)b * kN + n0 + r) * kD + d];
  }
  __syncthreads();
  const half_t* Kb = Kf + bb * kM * kD;
  const half_t* Vb = Vt + bb * kD * kM;
  const int rbase = (lane & 16) ? 8 : 0;
  const int cl = lane & 15;
  const float scale = 0.03125f;  // 1/sqrt(D)
  for (int round = 0; round < kTRAIL; ++round) {
    // scores = Q * K^T   [16 x 64]
    v8f s4[4] = {};
    for (int k0 = 0; k0 < kD; k0 += 32) {
      const v16h aF = frag_a(&Qh[cl][k0], lane);
#pragma unroll
      for (int mt = 0; mt < 4; ++mt) {
        const v16h bF = frag_b(Kb + (size_t)(mt * 16 + cl) * kD + k0, lane);
        s4[mt] = wmma_f16(aF, bF, s4[mt]);
      }
    }
#pragma unroll
    for (int mt = 0; mt < 4; ++mt)
#pragma unroll
      for (int r = 0; r < 8; ++r) sc[rbase + r][mt * 16 + cl] = s4[mt][r];
    __syncthreads();
    // softmax over 64 slots, one lane per row
    if (lane < 16) {
      float mx = -1e30f;
      for (int m = 0; m < kM; ++m) mx = fmaxf(mx, sc[lane][m]);
      float sum = 0.f;
      for (int m = 0; m < kM; ++m) {
        const float e_ = __expf((sc[lane][m] - mx) * scale);
        sc[lane][m] = e_;
        sum += e_;
      }
      const float is = 1.f / sum;
      for (int m = 0; m < kM; ++m) attnh[lane][m] = (half_t)(sc[lane][m] * is);
    }
    __syncthreads();
    // q = attn * V    [16 x 1024]
    for (int d0 = 0; d0 < kD; d0 += 16) {
      v8f acc = {};
#pragma unroll
      for (int kk = 0; kk < kM; kk += 32) {
        const v16h aF = frag_a(&attnh[cl][kk], lane);
        const v16h bF = frag_b(Vb + (size_t)(d0 + cl) * kM + kk, lane);
        acc = wmma_f16(aF, bF, acc);
      }
#pragma unroll
      for (int r = 0; r < 8; ++r) Qh[rbase + r][d0 + cl] = (half_t)acc[r];
    }
    __syncthreads();
  }
  for (int idx = lane; idx < 16 * kD; idx += 32) {
    const int r = idx >> 10, d = idx & (kD - 1);
    unsafeAtomicAdd(&integ[((size_t)b * kN + n0 + r) * kD + d], (float)Qh[r][d]);
  }
}

// ---------------- final LN ----------------

__global__ __launch_bounds__(256) void final_ln_kernel(const float* __restrict__ Hp,
                                                       const float* __restrict__ g,
                                                       const float* __restrict__ b,
                                                       half_t* __restrict__ outh) {
  __shared__ float red[256];
  const int bn = blockIdx.x;
  const int t = threadIdx.x;
  const size_t base = (size_t)bn * kD;
  float s = 0.f, s2 = 0.f;
  for (int d = t; d < kD; d += 256) {
    const float v = Hp[base + d];
    s += v;
    s2 += v * v;
  }
  const float sum = block_reduce_sum<256>(s, red);
  const float sq = block_reduce_sum<256>(s2, red);
  const float mu = sum * (1.f / kD);
  const float var = sq * (1.f / kD) - mu * mu;
  const float rstd = rsqrtf(var + 1e-5f);
  for (int d = t; d < kD; d += 256)
    outh[base + d] = (half_t)((Hp[base + d] - mu) * rstd * g[d] + b[d]);
}

__global__ void aux_kernel(const float* __restrict__ aux, float* __restrict__ out) {
  out[(size_t)kBS * kN * kVOCAB] = aux[0] * 0.1f / (float)((size_t)kBS * kN * kC * kDCOL);
}

// ---------------- host launch ----------------

extern "C" void kernel_launch(void* const* d_in, const int* in_sizes, int n_in,
                              void* d_out, int out_size, void* d_ws, size_t ws_size,
                              hipStream_t stream) {
  (void)in_sizes; (void)n_in; (void)out_size; (void)ws_size;
  const int* ids      = (const int*)d_in[0];
  const float* emb    = (const float*)d_in[1];
  const float* pos    = (const float*)d_in[2];
  const float* s1Win  = (const float*)d_in[3];
  const float* s1Wg   = (const float*)d_in[4];
  const float* s1Wout = (const float*)d_in[5];
  const float* s1g    = (const float*)d_in[6];
  const float* s1b    = (const float*)d_in[7];
  const float* s3Win  = (const float*)d_in[8];
  const float* s3Wg   = (const float*)d_in[9];
  const float* s3Wout = (const float*)d_in[10];
  const float* s3g    = (const float*)d_in[11];
  const float* s3b    = (const float*)d_in[12];
  const float* Wseed  = (const float*)d_in[13];
  const float* Ww     = (const float*)d_in[14];
  const float* Wpred  = (const float*)d_in[15];
  const float* Wnw    = (const float*)d_in[16];
  const float* Wnb    = (const float*)d_in[17];
  const float* Wdelta = (const float*)d_in[18];
  const float* pmstate= (const float*)d_in[19];
  const float* emK    = (const float*)d_in[20];
  const float* emV    = (const float*)d_in[21];
  const float* lng    = (const float*)d_in[22];
  const float* lnb    = (const float*)d_in[23];

  char* ws = (char*)d_ws;
  constexpr size_t MB = 1ull << 20;
  float*  X     = (float*)(ws + 0 * MB);     // [BS,N,D]      16 MB
  float*  XC    = (float*)(ws + 16 * MB);    // [BS,N,D]      16 MB
  float*  Z     = (float*)(ws + 32 * MB);    // [BS,N,C,E]    32 MB (shared: pm tmp / emb f16)
  float*  Aa    = (float*)(ws + 64 * MB);    // [BS,N,C,E]    32 MB
  float*  SURP  = (float*)(ws + 96 * MB);    // [BS,N,D]      16 MB
  float*  SEED  = (float*)(ws + 112 * MB);   // [BS,N,D]      16 MB
  float*  WCAND = (float*)(ws + 128 * MB);   // [BS,N,D]      16 MB
  float*  INTEG = (float*)(ws + 144 * MB);   // [BS,N,D]      16 MB
  half_t* SURPH = (half_t*)(ws + 160 * MB);  // f16           8 MB
  half_t* OUTH  = (half_t*)(ws + 168 * MB);  // f16           8 MB
  half_t* WDT   = (half_t*)(ws + 176 * MB);  // [NB,D,D] f16  16 MB
  half_t* KF    = (half_t*)(ws + 192 * MB);  // 4 MB
  half_t* KNH   = (half_t*)(ws + 196 * MB);  // 4 MB
  half_t* VT    = (half_t*)(ws + 200 * MB);  // 4 MB
  float*  NOVS  = (float*)(ws + 204 * MB);   // [BS,N] 16 KB
  float*  AUX   = (float*)(ws + 204 * MB + 65536);
  float*  TMPPM = Z;            // reused between stage-1 and stage-3 scans
  half_t* EMBH  = (half_t*)Z;   // reused after stage-3 (needs 62.5 MB of Z+Aa's 64 MB)

  float* logits = (float*)d_out;

  hipMemsetAsync(INTEG, 0, (size_t)kBS * kN * kD * sizeof(float), stream);
  hipMemsetAsync(AUX, 0, sizeof(float), stream);

  // weight / memory prep
  wdt_kernel<<<kNB * kD * kD / 256, 256, 0, stream>>>(Wdelta, WDT);
  kprep_kernel<<<kBS * kNB * kM, 256, 0, stream>>>(emK, KF, KNH);
  vt_kernel<<<kBS * kNB * kD * kM / 256, 256, 0, stream>>>(emV, VT);

  embed_kernel<<<kBS * kN, 256, 0, stream>>>(ids, emb, pos, X, XC);

  // stage 1 scan layers
  for (int l = 0; l < kL; ++l) {
    const size_t wo = (size_t)l * kC * kDCOL * kE;
    scan_phaseA<<<kBS * kN * kC, 128, 0, stream>>>(X, s1Win + wo, s1Wg + wo,
                                                   s1g + l * kDCOL, s1b + l * kDCOL, Z, Aa);
    scan_phaseB<<<kBS * kC * kE / 256, 256, 0, stream>>>(Z, Aa);
    scan_phaseC<<<kBS * kN * kC, 64, 0, stream>>>(X, Z, s1Wout + wo);
  }

  // PCM + seed/w_cand
  pcm_kernel<<<kBS * kN * kC, 64, 0, stream>>>(X, XC, Wpred, Wseed, Ww, SURP, SURPH,
                                               SEED, WCAND, AUX);

  // novelty (bank-summed)
  novelty_kernel<<<kBS * kN, 256, 0, stream>>>(X, WCAND, SURP, Wnw, Wnb, KNH, NOVS);

  // procedural memory: per-bank WMMA GEMM + causal gated read
  for (int bank = 0; bank < kNB; ++bank) {
    gemm_nt_kernel<kD><<<dim3(kD / 64, kBS * kN / 64), 128, 0, stream>>>(
        SURPH, WDT + (size_t)bank * kD * kD, TMPPM, kD);
    pm_cum_kernel<<<kBS * kD / 256, 256, 0, stream>>>(TMPPM, X, pmstate, INTEG, bank);
  }

  // episodic memory: trail attention (WMMA) + cumulative writes
  trail_attn_kernel<<<dim3(kN / 16, kNB, kBS), 32, 0, stream>>>(SEED, KF, VT, INTEG);
  em_cum_kernel<<<kBS * kD / 256, 256, 0, stream>>>(NOVS, WCAND, INTEG);

  // stage 3 scan layers (in place on INTEG)
  for (int l = 0; l < kL; ++l) {
    const size_t wo = (size_t)l * kC * kDCOL * kE;
    scan_phaseA<<<kBS * kN * kC, 128, 0, stream>>>(INTEG, s3Win + wo, s3Wg + wo,
                                                   s3g + l * kDCOL, s3b + l * kDCOL, Z, Aa);
    scan_phaseB<<<kBS * kC * kE / 256, 256, 0, stream>>>(Z, Aa);
    scan_phaseC<<<kBS * kN * kC, 64, 0, stream>>>(INTEG, Z, s3Wout + wo);
  }

  // final LN -> f16 activations
  final_ln_kernel<<<kBS * kN, 256, 0, stream>>>(INTEG, lng, lnb, OUTH);

  // tied LM head: logits = out @ emb^T  (WMMA)
  embh_kernel<<<kVOCAB * kD / 256, 256, 0, stream>>>(emb, EMBH);
  gemm_nt_kernel<kVOCAB><<<dim3(kVOCAB / 64, kBS * kN / 64), 128, 0, stream>>>(
      OUTH, EMBH, logits, kD);

  aux_kernel<<<1, 1, 0, stream>>>(AUX, logits);
}
